// MultiHeadedAttention_6682969113363
// MI455X (gfx1250) — matrix operand order repaired and checked by reading, not compiled, not run on hardware
//
#include <hip/hip_runtime.h>

// ---------------------------------------------------------------------------
// MI455X (gfx1250) multi-head attention, bf16 WMMA everywhere + TDM staging.
// B=2, S=2048, D=1024, H=16, DK=64
// ---------------------------------------------------------------------------

#define NB 2
#define NS 2048
#define ND 1024
#define NH 16
#define NDK 64

typedef __attribute__((ext_vector_type(16))) __bf16 v16bf;
typedef __attribute__((ext_vector_type(8)))  float  v8f;

#if defined(__gfx1250__) && __has_builtin(__builtin_amdgcn_tensor_load_to_lds) && \
    __has_builtin(__builtin_amdgcn_s_wait_tensorcnt)
#define USE_TDM 1
#endif

#ifdef USE_TDM
typedef __attribute__((ext_vector_type(4))) unsigned int u32x4;
typedef __attribute__((ext_vector_type(8))) int          i32x8;
typedef __attribute__((ext_vector_type(4))) int          i32x4;

// DMA a 128-row x 64-element (bf16) tile, row stride 64 elements in memory,
// into LDS with +16B padding per 128B row (row stride 72 elements in LDS).
// D# layout per CDNA5 ISA ch.8 (2-D tensor: groups 2/3 zero).
// This toolchain exposes the 6-arg builtin:
//   (u32x4 g0, i32x8 g1, i32x4 g2, i32x4 g3, i32x8 extra, i32 cpol)
static __device__ __forceinline__ void tdm_load_tile_128x64(const __bf16* gsrc,
                                                            __bf16* ldst) {
    unsigned long long ga = (unsigned long long)(uintptr_t)gsrc;
    unsigned int lds = (unsigned int)(uintptr_t)ldst;   // addr[31:0] = LDS offset
    u32x4 g0;
    g0[0] = 1u;                                   // count=1 (valid), user mode
    g0[1] = lds;                                  // lds_addr (bytes)
    g0[2] = (unsigned int)ga;                     // global_addr[31:0]
    g0[3] = (unsigned int)((ga >> 32) & 0x01FFFFFFull) | (2u << 30); // [56:32]|type=2
    i32x8 g1;
    g1[0] = (1 << 16)        // data_size = 2 bytes
          | (1 << 20)        // pad_enable
          | (4 << 22)        // pad_interval code 4 -> 32 dwords (128B) per pad
          | (3 << 25);       // pad_amount   code 3 -> 4 dwords (16B)
    g1[1] = (int)(64u << 16);     // tensor_dim0 = 64   (bits 79:48 low half)
    g1[2] = (int)(2048u << 16);   // tensor_dim1 = 2048 (bits 111:80 low half)
    g1[3] = (int)(64u << 16);     // tile_dim0 = 64     (bits 127:112)
    g1[4] = 128;                  // tile_dim1 = 128
    g1[5] = 64;                   // tensor_dim0_stride = 64 (bits 207:160 low)
    g1[6] = 0;
    g1[7] = 0;
    i32x4 z4 = {0, 0, 0, 0};
    i32x8 z8 = {0, 0, 0, 0, 0, 0, 0, 0};
    __builtin_amdgcn_tensor_load_to_lds(g0, g1, z4, z4, z8, 0);
}
#endif

struct U32x8 { uint4 lo, hi; };

// Build a 16-element bf16 fragment from two 16-byte LDS segments.
static __device__ __forceinline__ v16bf frag2(const __bf16* p0, const __bf16* p1) {
    U32x8 u;
    u.lo = *(const uint4*)p0;
    u.hi = *(const uint4*)p1;
    return __builtin_bit_cast(v16bf, u);
}

static __device__ __forceinline__ v8f vzero8() {
    v8f v;
#pragma unroll
    for (int i = 0; i < 8; ++i) v[i] = 0.0f;
    return v;
}

static __device__ __forceinline__ v8f wmma_bf16(v16bf a, v16bf b, v8f c) {
    return __builtin_amdgcn_wmma_f32_16x16x32_bf16(
        /*neg_a=*/false, a, /*neg_b=*/false, b,
        /*c_mod=*/(short)0, c, /*reuse_a=*/false, /*reuse_b=*/false);
}

// ---------------------------------------------------------------------------
// GEMM: C[4096,1024] = A[4096,1024] x W[1024,1024] + bias
//   AF32:  A is fp32 (converted to bf16 while staging), else A is bf16
//   SPLIT: output bf16 in split-head [B,H,S,DK] layout, else fp32 row-major
// 128x128 block tile, 8 waves of 32x64, K-step 32 (one WMMA K per step).
// ---------------------------------------------------------------------------
template <bool AF32, bool SPLIT>
__global__ __launch_bounds__(256) void gemm128(const void* __restrict__ Av,
                                               const float* __restrict__ W,
                                               const float* __restrict__ bias,
                                               void* __restrict__ Cv) {
    __shared__ __bf16 sA[128 * 40];   // [row][k] stride 40 (pad)
    __shared__ __bf16 sB[128 * 40];   // [n][k]   stride 40 (pad) = W tile transposed

    const int tid  = threadIdx.x;
    const int lane = tid & 31;
    const int w    = tid >> 5;   // 0..7
    const int wm   = w >> 1;     // 0..3 (M)
    const int wn   = w & 1;      // 0..1 (N)
    const int m0   = blockIdx.y * 128;
    const int n0   = blockIdx.x * 128;
    const int l16  = lane & 15;
    const int lh   = lane >> 4;

    v8f acc[2][4];
#pragma unroll
    for (int i = 0; i < 2; ++i)
#pragma unroll
        for (int j = 0; j < 4; ++j) acc[i][j] = vzero8();

    for (int kk = 0; kk < ND; kk += 32) {
        // ---- stage A tile (128x32) ----
        if constexpr (AF32) {
            const float* A = (const float*)Av;
#pragma unroll
            for (int p = 0; p < 4; ++p) {
                int row = (tid >> 3) + p * 32;
                int col = (tid & 7) * 4;
                float4 v = *(const float4*)&A[(size_t)(m0 + row) * ND + kk + col];
                __bf16* d = &sA[row * 40 + col];
                d[0] = (__bf16)v.x; d[1] = (__bf16)v.y;
                d[2] = (__bf16)v.z; d[3] = (__bf16)v.w;
            }
        } else {
            const __bf16* A = (const __bf16*)Av;
#pragma unroll
            for (int p = 0; p < 2; ++p) {
                int row = (tid >> 2) + p * 64;
                int seg = (tid & 3) * 8;
                *(uint4*)&sA[row * 40 + seg] =
                    *(const uint4*)&A[(size_t)(m0 + row) * ND + kk + seg];
            }
        }
        // ---- stage W tile (32x128) transposed into sB[n][k] ----
#pragma unroll
        for (int p = 0; p < 4; ++p) {
            int k = (tid >> 5) + p * 8;
            int n = (tid & 31) * 4;
            float4 v = *(const float4*)&W[(size_t)(kk + k) * ND + n0 + n];
            sB[(n + 0) * 40 + k] = (__bf16)v.x;
            sB[(n + 1) * 40 + k] = (__bf16)v.y;
            sB[(n + 2) * 40 + k] = (__bf16)v.z;
            sB[(n + 3) * 40 + k] = (__bf16)v.w;
        }
        // prefetch next W tile into the cache hierarchy (global_prefetch_b8)
        if (kk + 32 < ND)
            __builtin_prefetch(&W[(size_t)(kk + 32 + (tid >> 5)) * ND + n0 + (tid & 31) * 4], 0, 0);
        __syncthreads();

        v16bf af[2], bf[4];
#pragma unroll
        for (int i = 0; i < 2; ++i) {
            int m  = wm * 32 + i * 16 + l16;
            int kh = lh * 8;
            af[i] = frag2(&sA[m * 40 + kh], &sA[m * 40 + kh + 16]);
        }
#pragma unroll
        for (int j = 0; j < 4; ++j) {
            int n  = wn * 64 + j * 16 + l16;
            int kb = lh * 8;
            bf[j] = frag2(&sB[n * 40 + kb], &sB[n * 40 + kb + 16]);
        }
#pragma unroll
        for (int i = 0; i < 2; ++i)
#pragma unroll
            for (int j = 0; j < 4; ++j)
                acc[i][j] = wmma_bf16(af[i], bf[j], acc[i][j]);
        __syncthreads();
    }

    // ---- epilogue (C layout: VGPR r -> row r / r+8 by lane half, N = lane%16) ----
#pragma unroll
    for (int i = 0; i < 2; ++i) {
#pragma unroll
        for (int j = 0; j < 4; ++j) {
            int   gcol = n0 + wn * 64 + j * 16 + l16;
            float bv   = bias[gcol];
#pragma unroll
            for (int r = 0; r < 8; ++r) {
                int   grow = m0 + wm * 32 + i * 16 + r + 8 * lh;
                float v    = acc[i][j][r] + bv;
                if constexpr (SPLIT) {
                    int b = grow >> 11, s = grow & (NS - 1);
                    int h = gcol >> 6, dk = gcol & (NDK - 1);
                    ((__bf16*)Cv)[((size_t)(b * NH + h) * NS + s) * NDK + dk] = (__bf16)v;
                } else {
                    ((float*)Cv)[(size_t)grow * ND + gcol] = v;
                }
            }
        }
    }
}

// ---------------------------------------------------------------------------
// Flash attention: one block = 128 queries of one (b,h); loop over 128-key
// tiles. 8 waves x 16 queries each. Online softmax via wave32 xor-shuffles.
// Q and K tiles staged by the Tensor Data Mover (wave 0 issues the DMA and
// waits on TENSORcnt); V (needs 16-bit transpose) and mask (int32 -> byte)
// staged by all waves, overlapping the DMA.
// Dynamic LDS layout (bytes):
//   sK : [128][72] bf16              = 18432
//   sV : [64][136] bf16 (V^T)        = 17408
//   sM : [128][128] u8               = 16384
//   sP : 8 waves x [16][136] bf16    = 34816   (aliased by sQ in prologue)
// ---------------------------------------------------------------------------
#define ATT_SMEM (18432 + 17408 + 16384 + 34816)

__global__ __launch_bounds__(256) void attn128(const __bf16* __restrict__ Q,
                                               const __bf16* __restrict__ K,
                                               const __bf16* __restrict__ V,
                                               const int* __restrict__ mask,
                                               __bf16* __restrict__ merged) {
    extern __shared__ char smem[];
    __bf16*        sK = (__bf16*)smem;
    __bf16*        sV = (__bf16*)(smem + 18432);
    unsigned char* sM = (unsigned char*)(smem + 18432 + 17408);
    __bf16*        sP = (__bf16*)(smem + 18432 + 17408 + 16384);
    __bf16*        sQ = sP;   // prologue-only, safely aliased

    const int tid  = threadIdx.x;
    const int lane = tid & 31;
    const int w    = tid >> 5;
    const int l16  = lane & 15;
    const int lh   = lane >> 4;
    const int bh   = blockIdx.y;          // b*H + h
    const int b    = bh >> 4;
    const int h    = bh & 15;
    const int q0   = blockIdx.x * 128;

    const __bf16* Qg = Q + (size_t)bh * NS * NDK;
    const __bf16* Kg = K + (size_t)bh * NS * NDK;
    const __bf16* Vg = V + (size_t)bh * NS * NDK;

    // ---- stage Q tile [128][64] (stride 72) and lift fragments to registers ----
#ifdef USE_TDM
    if (w == 0) {
        tdm_load_tile_128x64(Qg + (size_t)q0 * NDK, sQ);
        __builtin_amdgcn_s_wait_tensorcnt(0);
    }
#else
#pragma unroll
    for (int p = 0; p < 4; ++p) {
        int row = (tid >> 3) + p * 32;
        int seg = (tid & 7) * 8;
        *(uint4*)&sQ[row * 72 + seg] =
            *(const uint4*)&Qg[(size_t)(q0 + row) * NDK + seg];
    }
#endif
    __syncthreads();
    v16bf qf[2];
#pragma unroll
    for (int ks = 0; ks < 2; ++ks) {
        int m  = w * 16 + l16;
        int kh = ks * 32 + lh * 8;
        qf[ks] = frag2(&sQ[m * 72 + kh], &sQ[m * 72 + kh + 16]);
    }

    float mrow[8], lrow[8];
    v8f   o[4];
#pragma unroll
    for (int r = 0; r < 8; ++r) { mrow[r] = -1e30f; lrow[r] = 0.0f; }
#pragma unroll
    for (int a = 0; a < 4; ++a) o[a] = vzero8();

    __bf16* sPw = sP + w * 16 * 136;

    for (int t = 0; t < NS / 128; ++t) {
        const int k0 = t * 128;
        __syncthreads();   // t==0: all waves done reading aliased sQ; else prior-iter reads

        // ---- stage K tile [128][64] row-major, stride 72 ----
#ifdef USE_TDM
        if (w == 0) tdm_load_tile_128x64(Kg + (size_t)k0 * NDK, sK);
#else
#pragma unroll
        for (int p = 0; p < 4; ++p) {
            int row = (tid >> 3) + p * 32;
            int seg = (tid & 7) * 8;
            *(uint4*)&sK[row * 72 + seg] =
                *(const uint4*)&Kg[(size_t)(k0 + row) * NDK + seg];
        }
#endif
        // ---- stage V tile transposed -> sV[dk][key], stride 136 ----
#pragma unroll
        for (int p = 0; p < 4; ++p) {
            int kk2 = (tid >> 3) + p * 32;
            int dk0 = (tid & 7) * 8;
            uint4  vv = *(const uint4*)&Vg[(size_t)(k0 + kk2) * NDK + dk0];
            __bf16 tv[8];
            *(uint4*)tv = vv;
#pragma unroll
            for (int i = 0; i < 8; ++i) sV[(dk0 + i) * 136 + kk2] = tv[i];
        }
        // ---- stage mask tile as bytes ----
#pragma unroll
        for (int p = 0; p < 16; ++p) {
            int idx = tid + p * 256;       // int4 granules over 128x128
            int q   = idx >> 5;
            int kq  = (idx & 31) * 4;
            int4 mv = *(const int4*)&mask[((size_t)(b * NS + q0 + q)) * NS + k0 + kq];
            unsigned int packed = (mv.x ? 1u : 0u) | ((mv.y ? 1u : 0u) << 8) |
                                  ((mv.z ? 1u : 0u) << 16) | ((mv.w ? 1u : 0u) << 24);
            *(unsigned int*)&sM[q * 128 + kq] = packed;
        }
#ifdef USE_TDM
        if (w == 0) __builtin_amdgcn_s_wait_tensorcnt(0);
#endif
        __syncthreads();

        // ---- scores: 16 queries x 128 keys per wave (8 accumulators) ----
        v8f sc[8];
#pragma unroll
        for (int t8 = 0; t8 < 8; ++t8) {
            sc[t8] = vzero8();
#pragma unroll
            for (int ks = 0; ks < 2; ++ks) {
                int n  = t8 * 16 + l16;            // key column
                int kb = ks * 32 + lh * 8;         // dk chunk
                v16bf kf = frag2(&sK[n * 72 + kb], &sK[n * 72 + kb + 16]);
                sc[t8]   = wmma_bf16(qf[ks], kf, sc[t8]);
            }
        }

        // ---- online softmax: row max via xor-shuffles (halves independent) ----
        const float scale = 0.125f;   // 1/sqrt(64)
        float corr[8];
#pragma unroll
        for (int r = 0; r < 8; ++r) {
            float mx = mrow[r];
#pragma unroll
            for (int t8 = 0; t8 < 8; ++t8) {
                float sv = sc[t8][r] * scale;
                sc[t8][r] = sv;
                mx = fmaxf(mx, sv);
            }
            mx = fmaxf(mx, __shfl_xor(mx, 1));
            mx = fmaxf(mx, __shfl_xor(mx, 2));
            mx = fmaxf(mx, __shfl_xor(mx, 4));
            mx = fmaxf(mx, __shfl_xor(mx, 8));
            corr[r] = __expf(mrow[r] - mx);
            mrow[r] = mx;
            lrow[r] *= corr[r];
        }
#pragma unroll
        for (int a = 0; a < 4; ++a)
#pragma unroll
            for (int r = 0; r < 8; ++r) o[a][r] *= corr[r];

        // ---- P = exp(s-m)*mask, row sums, restage P to LDS in A layout ----
#pragma unroll
        for (int r = 0; r < 8; ++r) {
            int   qloc = w * 16 + r + 8 * lh;
            float rs   = 0.0f;
#pragma unroll
            for (int t8 = 0; t8 < 8; ++t8) {
                float pv = __expf(sc[t8][r] - mrow[r]);
                pv *= (float)sM[qloc * 128 + t8 * 16 + l16];
                rs += pv;
                sPw[(r + 8 * lh) * 136 + t8 * 16 + l16] = (__bf16)pv;
            }
            rs += __shfl_xor(rs, 1);
            rs += __shfl_xor(rs, 2);
            rs += __shfl_xor(rs, 4);
            rs += __shfl_xor(rs, 8);
            lrow[r] += rs;
        }

        // ---- O += P x V (K-dim = 128 keys = 4 WMMA steps) ----
        v16bf pf[4];
#pragma unroll
        for (int kst = 0; kst < 4; ++kst) {
            int kh = kst * 32 + lh * 8;
            pf[kst] = frag2(&sPw[l16 * 136 + kh], &sPw[l16 * 136 + kh + 16]);
        }
#pragma unroll
        for (int a = 0; a < 4; ++a) {
#pragma unroll
            for (int kst = 0; kst < 4; ++kst) {
                int n  = a * 16 + l16;             // dk column
                int kb = kst * 32 + lh * 8;        // key chunk
                v16bf vf = frag2(&sV[n * 136 + kb], &sV[n * 136 + kb + 16]);
                o[a]     = wmma_bf16(pf[kst], vf, o[a]);
            }
        }
    }

    // ---- normalize and store merged [B,S,D] bf16 ----
#pragma unroll
    for (int r = 0; r < 8; ++r) {
        float inv = lrow[r] > 0.0f ? 1.0f / lrow[r] : 0.0f;
        int   q   = q0 + w * 16 + r + 8 * lh;
#pragma unroll
        for (int a = 0; a < 4; ++a) {
            int col = h * NDK + a * 16 + l16;
            merged[((size_t)(b * NS + q)) * ND + col] = (__bf16)(o[a][r] * inv);
        }
    }
}

// ---------------------------------------------------------------------------
// Host launcher
// ---------------------------------------------------------------------------
extern "C" void kernel_launch(void* const* d_in, const int* in_sizes, int n_in,
                              void* d_out, int out_size, void* d_ws, size_t ws_size,
                              hipStream_t stream) {
    const float* query = (const float*)d_in[0];
    const float* key   = (const float*)d_in[1];
    const float* value = (const float*)d_in[2];
    const int*   mask  = (const int*)d_in[3];
    const float* Wq = (const float*)d_in[4];
    const float* bq = (const float*)d_in[5];
    const float* Wk = (const float*)d_in[6];
    const float* bk = (const float*)d_in[7];
    const float* Wv = (const float*)d_in[8];
    const float* bv = (const float*)d_in[9];
    const float* Wo = (const float*)d_in[10];
    const float* bo = (const float*)d_in[11];

    const size_t nelem = (size_t)NB * NS * ND;   // 4,194,304
    char* ws = (char*)d_ws;                       // needs 4 * nelem * 2 = 32 MB
    __bf16* Qb = (__bf16*)(ws + nelem * 0);
    __bf16* Kb = (__bf16*)(ws + nelem * 2);
    __bf16* Vb = (__bf16*)(ws + nelem * 4);
    __bf16* Mg = (__bf16*)(ws + nelem * 6);

    dim3 gg(ND / 128, (NB * NS) / 128);           // 8 x 32
    gemm128<true, true><<<gg, 256, 0, stream>>>(query, Wq, bq, Qb);
    gemm128<true, true><<<gg, 256, 0, stream>>>(key,   Wk, bk, Kb);
    gemm128<true, true><<<gg, 256, 0, stream>>>(value, Wv, bv, Vb);

    dim3 ga(NS / 128, NB * NH);                   // 16 x 32
    attn128<<<ga, 256, ATT_SMEM, stream>>>(Qb, Kb, Vb, mask, Mg);

    gemm128<false, false><<<gg, 256, 0, stream>>>(Mg, Wo, bo, d_out);
}
